// GATNetwork_1357209666144
// MI455X (gfx1250) — matrix operand order repaired and checked by reading, not compiled, not run on hardware
//
#include <hip/hip_runtime.h>
#include <hip/hip_bf16.h>

// ---------------------------------------------------------------------------
// Vector types for CDNA5 WMMA
// ---------------------------------------------------------------------------
typedef __attribute__((ext_vector_type(16))) __bf16 v16bf;
typedef __attribute__((ext_vector_type(8)))  __bf16 v8bf;
typedef __attribute__((ext_vector_type(8)))  float  v8f;

__device__ inline __bf16 f2bf(float f) {
    unsigned u = __builtin_bit_cast(unsigned, f);
    unsigned r = (u + 0x7FFFu + ((u >> 16) & 1u)) >> 16;   // RNE bf16
    unsigned short s = (unsigned short)r;
    return __builtin_bit_cast(__bf16, s);
}

// ---------------------------------------------------------------------------
// fp32 -> bf16 elementwise (x8 vectorized)
// ---------------------------------------------------------------------------
__global__ void cvt_f32_to_bf16_x8(const float* __restrict__ s,
                                   __bf16* __restrict__ d, long long n) {
    long long i = ((long long)blockIdx.x * blockDim.x + threadIdx.x) * 8;
    if (i >= n) return;
    if (i + 8 <= n) {
        v8bf o;
#pragma unroll
        for (int j = 0; j < 8; ++j) o[j] = f2bf(s[i + j]);
        *(v8bf*)(d + i) = o;
    } else {
        for (long long j = i; j < n; ++j) d[j] = f2bf(s[j]);
    }
}

// W[Kin,Kout] fp32 -> WT[Kout,Kin] bf16 (transpose + convert; weights small)
__global__ void wt_transpose_bf16(const float* __restrict__ W,
                                  __bf16* __restrict__ WT, int Kin, int Kout) {
    int idx = blockIdx.x * blockDim.x + threadIdx.x;
    if (idx >= Kin * Kout) return;
    int k = idx / Kout;
    int c = idx - k * Kout;
    WT[(size_t)c * Kin + k] = f2bf(W[idx]);
}

// ---------------------------------------------------------------------------
// GEMM: C[N,Kout] = A[N,Kin] @ W[Kin,Kout]; A bf16 row-major, WT bf16 [Kout,Kin].
// Block: 256 threads = 8 waves; each wave owns one 16-row M-tile and NT 16-col
// N-tiles (NT*16 columns per block; all 8 waves share the LDS weight panel).
// Fragment layouts used here (wave32; contiguous K, same order for A and B, not the ISA table order):
//   A (16x32): lanes 0-15 row=lane, K=0..15; lanes 16-31 K=16..31
//   B (32x16): lanes 0-15 col=lane K=0..15; lanes 16-31 col=lane-16 K=16..31
//   C/D: lane<16 col=lane rows 0..7; lane>=16 col=lane-16 rows 8..15
// Inner loop: batch all NT B-fragment LDS loads (one ds clause + single
// s_wait_dscnt), then issue NT back-to-back WMMAs for XDL pipelining.
// ---------------------------------------------------------------------------
template <int NT>
__global__ void gemm_bf16_wmma(const __bf16* __restrict__ A,
                               const __bf16* __restrict__ WT,
                               float* __restrict__ C,
                               int Kin, int Kout, int Mtiles) {
    extern __shared__ __bf16 lds[];            // (NT*16) x Kin panel of WT
    const int COLS = NT * 16;
    const int col0 = blockIdx.y * COLS;

    // Cooperative load of weight panel into LDS (contiguous per column row).
    {
        int chunks = (COLS * Kin) >> 3;        // 16-byte chunks
        for (int c = threadIdx.x; c < chunks; c += blockDim.x) {
            int off = c << 3;
            int cc = off / Kin;
            int kk = off - cc * Kin;
            *(v8bf*)(lds + off) =
                *(const v8bf*)(WT + (size_t)(col0 + cc) * Kin + kk);
        }
    }
    __syncthreads();

    int wave = threadIdx.x >> 5;
    int mtile = blockIdx.x * 8 + wave;
    if (mtile >= Mtiles) return;               // wave-uniform: EXEC stays all-1

    int lane = threadIdx.x & 31;
    int half = lane >> 4;
    int l15  = lane & 15;
    const int akb = half ? 16 : 0;             // A K sub-offset for lane half
    const int bko = half ? 16 : 0;             // B K offset for lane half

    const __bf16* arow = A + ((size_t)(mtile * 16 + l15)) * Kin;

    v8f zero = {};
    v8f acc[NT];
#pragma unroll
    for (int t = 0; t < NT; ++t) acc[t] = zero;

    for (int k0 = 0; k0 < Kin; k0 += 32) {
        // Prefetch next A chunk toward the WGP while this step computes.
        __builtin_prefetch(arow + k0 + 32, 0, 0);

        v8bf alo = *(const v8bf*)(arow + k0 + akb);
        v8bf ahi = *(const v8bf*)(arow + k0 + akb + 8);
        v16bf a = __builtin_shufflevector(alo, ahi, 0, 1, 2, 3, 4, 5, 6, 7,
                                          8, 9, 10, 11, 12, 13, 14, 15);

        v16bf b[NT];
#pragma unroll
        for (int t = 0; t < NT; ++t) {          // batched ds_load clause
            const __bf16* bcol = lds + (size_t)(t * 16 + l15) * Kin + k0 + bko;
            v8bf blo = *(const v8bf*)(bcol);
            v8bf bhi = *(const v8bf*)(bcol + 8);
            b[t] = __builtin_shufflevector(blo, bhi, 0, 1, 2, 3, 4, 5, 6, 7,
                                           8, 9, 10, 11, 12, 13, 14, 15);
        }
#pragma unroll
        for (int t = 0; t < NT; ++t) {          // back-to-back WMMA chain
            acc[t] = __builtin_amdgcn_wmma_f32_16x16x32_bf16(
                false, a, false, b[t], (short)0, acc[t], false, false);
        }
    }

#pragma unroll
    for (int t = 0; t < NT; ++t) {
        float* crow = C + (size_t)(mtile * 16 + half * 8) * Kout +
                      col0 + t * 16 + l15;
#pragma unroll
        for (int v = 0; v < 8; ++v) crow[(size_t)v * Kout] = acc[t][v];
    }
}

// ---------------------------------------------------------------------------
// Utility fill
// ---------------------------------------------------------------------------
__global__ void fill_f32(float* __restrict__ p, long long n, float v) {
    long long i = (long long)blockIdx.x * blockDim.x + threadIdx.x;
    if (i < n) p[i] = v;
}

// float atomic max via int/uint atomics (mixed-sign safe; init -inf)
__device__ inline void atomicMaxFloat(float* addr, float val) {
    if (val >= 0.0f) {
        atomicMax((int*)addr, __float_as_int(val));
    } else {
        atomicMin((unsigned int*)addr, __float_as_uint(val));
    }
}

// ---------------------------------------------------------------------------
// Edge kernels. e < E -> (src=ei[e], dst=ei[E+e]); e >= E -> self loop.
// One thread per (edge, head). xl/xr: [N,H,D] f32. att: [H,D].
// ---------------------------------------------------------------------------
__global__ void edge_score(const int* __restrict__ ei, int E, int N, int H, int D,
                           const float* __restrict__ xl,
                           const float* __restrict__ xr,
                           const float* __restrict__ att,
                           float* __restrict__ scores,
                           float* __restrict__ m) {
    long long t = (long long)blockIdx.x * blockDim.x + threadIdx.x;
    long long tot = (long long)(E + N) * H;
    if (t >= tot) return;
    int e = (int)(t / H);
    int h = (int)(t - (long long)e * H);
    int s, d;
    if (e < E) { s = ei[e]; d = ei[E + e]; } else { s = d = e - E; }
    const float4* pl = (const float4*)(xl + ((size_t)s * H + h) * D);
    const float4* pr = (const float4*)(xr + ((size_t)d * H + h) * D);
    const float4* pa = (const float4*)(att + (size_t)h * D);
    float acc = 0.0f;
    int q = D >> 2;
    for (int i = 0; i < q; ++i) {
        float4 a = pl[i], b = pr[i], w = pa[i];
        float g0 = a.x + b.x, g1 = a.y + b.y, g2 = a.z + b.z, g3 = a.w + b.w;
        g0 = g0 > 0.f ? g0 : 0.2f * g0;
        g1 = g1 > 0.f ? g1 : 0.2f * g1;
        g2 = g2 > 0.f ? g2 : 0.2f * g2;
        g3 = g3 > 0.f ? g3 : 0.2f * g3;
        acc = fmaf(g0, w.x, acc);
        acc = fmaf(g1, w.y, acc);
        acc = fmaf(g2, w.z, acc);
        acc = fmaf(g3, w.w, acc);
    }
    scores[(size_t)e * H + h] = acc;
    atomicMaxFloat(m + (size_t)d * H + h, acc);
}

__global__ void edge_expsum(const int* __restrict__ ei, int E, int N, int H,
                            const float* __restrict__ m,
                            float* __restrict__ scores,
                            float* __restrict__ ssum) {
    long long t = (long long)blockIdx.x * blockDim.x + threadIdx.x;
    long long tot = (long long)(E + N) * H;
    if (t >= tot) return;
    int e = (int)(t / H);
    int h = (int)(t - (long long)e * H);
    int d;
    if (e < E) { d = ei[E + e]; } else { d = e - E; }
    size_t idx = (size_t)e * H + h;
    float ex = __expf(scores[idx] - m[(size_t)d * H + h]);
    scores[idx] = ex;
    atomicAdd(ssum + (size_t)d * H + h, ex);
}

__global__ void edge_aggr(const int* __restrict__ ei, int E, int N, int H, int D,
                          const float* __restrict__ xl,
                          const float* __restrict__ scores,
                          const float* __restrict__ ssum,
                          float* __restrict__ out) {
    long long t = (long long)blockIdx.x * blockDim.x + threadIdx.x;
    long long tot = (long long)(E + N) * H;
    if (t >= tot) return;
    int e = (int)(t / H);
    int h = (int)(t - (long long)e * H);
    int s, d;
    if (e < E) { s = ei[e]; d = ei[E + e]; } else { s = d = e - E; }
    float alpha = scores[(size_t)e * H + h] /
                  (ssum[(size_t)d * H + h] + 1e-16f);
    const float4* pl = (const float4*)(xl + ((size_t)s * H + h) * D);
    float* po = out + ((size_t)d * H + h) * D;
    int q = D >> 2;
    for (int i = 0; i < q; ++i) {
        float4 a = pl[i];
        atomicAdd(po + i * 4 + 0, alpha * a.x);
        atomicAdd(po + i * 4 + 1, alpha * a.y);
        atomicAdd(po + i * 4 + 2, alpha * a.z);
        atomicAdd(po + i * 4 + 3, alpha * a.w);
    }
}

// h[i] = elu(h[i] + b[i % hid])
__global__ void bias_elu(float* __restrict__ h, const float* __restrict__ b,
                         long long n, int hid) {
    long long i = (long long)blockIdx.x * blockDim.x + threadIdx.x;
    if (i >= n) return;
    float v = h[i] + b[(int)(i % hid)];
    h[i] = v > 0.0f ? v : (__expf(v) - 1.0f);
}

// out[n,:] = log_softmax(agg[n,:] + b), 16 classes
__global__ void bias_logsoftmax16(const float* __restrict__ agg,
                                  const float* __restrict__ b,
                                  float* __restrict__ out, int N) {
    int n = blockIdx.x * blockDim.x + threadIdx.x;
    if (n >= N) return;
    float v[16];
    float mx = -1e30f;
#pragma unroll
    for (int c = 0; c < 16; ++c) {
        v[c] = agg[(size_t)n * 16 + c] + b[c];
        mx = fmaxf(mx, v[c]);
    }
    float s = 0.0f;
#pragma unroll
    for (int c = 0; c < 16; ++c) s += __expf(v[c] - mx);
    float ls = __logf(s) + mx;
#pragma unroll
    for (int c = 0; c < 16; ++c) out[(size_t)n * 16 + c] = v[c] - ls;
}

// ---------------------------------------------------------------------------
// Host-side orchestration
// ---------------------------------------------------------------------------
extern "C" void kernel_launch(void* const* d_in, const int* in_sizes, int n_in,
                              void* d_out, int out_size, void* d_ws, size_t ws_size,
                              hipStream_t stream) {
    const float* x  = (const float*)d_in[0];
    const int*   ei = (const int*)d_in[1];
    const float* Wl[4] = {(const float*)d_in[2],  (const float*)d_in[6],
                          (const float*)d_in[10], (const float*)d_in[14]};
    const float* Wr[4] = {(const float*)d_in[3],  (const float*)d_in[7],
                          (const float*)d_in[11], (const float*)d_in[15]};
    const float* At[4] = {(const float*)d_in[4],  (const float*)d_in[8],
                          (const float*)d_in[12], (const float*)d_in[16]};
    const float* Bs[4] = {(const float*)d_in[5],  (const float*)d_in[9],
                          (const float*)d_in[13], (const float*)d_in[17]};
    float* outp = (float*)d_out;

    const int Fin = 128, HID = 256, Hh = 8, Dd = 32;
    const int N = in_sizes[0] / Fin;     // 50000
    const int E = in_sizes[1] / 2;       // 800000
    const int Mtiles = N / 16;           // 3125 (exact)

    // Workspace layout
    float* ws = (float*)d_ws;
    float* xl   = ws;                                  // N*HID
    float* xr   = xl + (size_t)N * HID;                // N*HID
    float* mbuf = xr + (size_t)N * HID;                // N*Hh
    float* sbuf = mbuf + (size_t)N * Hh;               // N*Hh
    float* sc   = sbuf + (size_t)N * Hh;               // (E+N)*Hh
    float* hA   = sc + (size_t)(E + N) * Hh;           // N*HID
    float* hB   = hA + (size_t)N * HID;                // N*HID
    __bf16* Abf = (__bf16*)(hB + (size_t)N * HID);     // N*HID bf16
    __bf16* WTb = Abf + (size_t)N * HID;               // HID*HID bf16

    const float NEG_INF = -__builtin_huge_valf();

    auto fill = [&](float* p, long long n, float v) {
        int blocks = (int)((n + 255) / 256);
        hipLaunchKernelGGL(fill_f32, dim3(blocks), dim3(256), 0, stream, p, n, v);
    };
    auto cvtA = [&](const float* A, int kin) {
        long long n = (long long)N * kin;
        int blocks = (int)((n / 8 + 255) / 256);
        hipLaunchKernelGGL(cvt_f32_to_bf16_x8, dim3(blocks), dim3(256), 0, stream,
                           A, Abf, n);
    };
    auto gemm = [&](const float* W, float* C, int kin, int kout) {
        int wn = kin * kout;
        hipLaunchKernelGGL(wt_transpose_bf16, dim3((wn + 255) / 256), dim3(256),
                           0, stream, W, WTb, kin, kout);
        if (kout >= 64) {
            dim3 grid((Mtiles + 7) / 8, kout / 64);
            size_t shmem = (size_t)64 * kin * sizeof(__bf16);
            hipLaunchKernelGGL((gemm_bf16_wmma<4>), grid, dim3(256), shmem,
                               stream, Abf, WTb, C, kin, kout, Mtiles);
        } else {
            dim3 grid((Mtiles + 7) / 8, kout / 16);
            size_t shmem = (size_t)16 * kin * sizeof(__bf16);
            hipLaunchKernelGGL((gemm_bf16_wmma<1>), grid, dim3(256), shmem,
                               stream, Abf, WTb, C, kin, kout, Mtiles);
        }
    };
    auto edge_layer = [&](const float* att, float* agg, int H, int D) {
        long long tot = (long long)(E + N) * H;
        int eb = (int)((tot + 255) / 256);
        fill(mbuf, (long long)N * H, NEG_INF);
        fill(sbuf, (long long)N * H, 0.0f);
        fill(agg, (long long)N * H * D, 0.0f);
        hipLaunchKernelGGL(edge_score, dim3(eb), dim3(256), 0, stream,
                           ei, E, N, H, D, xl, xr, att, sc, mbuf);
        hipLaunchKernelGGL(edge_expsum, dim3(eb), dim3(256), 0, stream,
                           ei, E, N, H, mbuf, sc, sbuf);
        hipLaunchKernelGGL(edge_aggr, dim3(eb), dim3(256), 0, stream,
                           ei, E, N, H, D, xl, sc, sbuf, agg);
    };
    auto elu = [&](float* h, const float* b, int hid) {
        long long n = (long long)N * hid;
        int blocks = (int)((n + 255) / 256);
        hipLaunchKernelGGL(bias_elu, dim3(blocks), dim3(256), 0, stream,
                           h, b, n, hid);
    };

    // ---- Layer 1: 128 -> 256, H=8, D=32 -> hA
    cvtA(x, Fin);
    gemm(Wl[0], xl, Fin, HID);
    gemm(Wr[0], xr, Fin, HID);
    edge_layer(At[0], hA, Hh, Dd);
    elu(hA, Bs[0], HID);

    // ---- Layer 2: 256 -> 256 -> hB
    cvtA(hA, HID);
    gemm(Wl[1], xl, HID, HID);
    gemm(Wr[1], xr, HID, HID);
    edge_layer(At[1], hB, Hh, Dd);
    elu(hB, Bs[1], HID);

    // ---- Layer 3: 256 -> 256 -> hA
    cvtA(hB, HID);
    gemm(Wl[2], xl, HID, HID);
    gemm(Wr[2], xr, HID, HID);
    edge_layer(At[2], hA, Hh, Dd);
    elu(hA, Bs[2], HID);

    // ---- Layer 4: 256 -> 16, H=1, D=16 -> hB (uses N*16 of it)
    cvtA(hA, HID);
    gemm(Wl[3], xl, HID, 16);
    gemm(Wr[3], xr, HID, 16);
    edge_layer(At[3], hB, 1, 16);

    // ---- Final: bias + log_softmax
    {
        int blocks = (N + 255) / 256;
        hipLaunchKernelGGL(bias_logsoftmax16, dim3(blocks), dim3(256), 0, stream,
                           hB, Bs[3], outp, N);
    }
}
